// SpatioTemporalInteractionv3_51762945852131
// MI455X (gfx1250) — matrix operand order repaired and checked by reading, not compiled, not run on hardware
//
#include <hip/hip_runtime.h>

typedef __attribute__((ext_vector_type(16))) _Float16 v16h;
typedef __attribute__((ext_vector_type(8)))  _Float16 v8h;
typedef __attribute__((ext_vector_type(8)))  float    v8f;
typedef __attribute__((ext_vector_type(4)))  unsigned int u32x4;
typedef __attribute__((ext_vector_type(8)))  int      i32x8;
typedef __attribute__((ext_vector_type(4)))  int      i32x4;

#define T_STEPS 32
#define B_DIM   64
#define N_DIM   2048
#define TB      (T_STEPS * B_DIM)        // 2048
#define BN      (B_DIM * N_DIM)          // 131072

// LIF step: v = 0.5v + x ; spike at v>=1 ; hard reset
__device__ __forceinline__ float lif_step(float& v, float x) {
  v = 0.5f * v + x;
  float s = (v >= 1.0f) ? 1.0f : 0.0f;
  v = (1.0f - s) * v;
  return s;
}

// ---------------- reductions ----------------
__global__ void k_spatial(const float* __restrict__ x, float* __restrict__ ssum) {
  __shared__ float red[256];
  const int tb = blockIdx.x;
  const float* row = x + (size_t)tb * N_DIM;
  float s = 0.f;
  for (int i = threadIdx.x; i < N_DIM; i += 256) s += row[i];
  red[threadIdx.x] = s;
  __syncthreads();
  for (int off = 128; off > 0; off >>= 1) {
    if (threadIdx.x < off) red[threadIdx.x] += red[threadIdx.x + off];
    __syncthreads();
  }
  if (threadIdx.x == 0) ssum[tb] = red[0];
}

__global__ void k_temporal(const float* __restrict__ x, float* __restrict__ tsum) {
  const int bn = blockIdx.x * blockDim.x + threadIdx.x;
  float s = 0.f;
  #pragma unroll
  for (int t = 0; t < T_STEPS; ++t) s += x[(size_t)t * BN + bn];
  tsum[bn] = s;
}

// ---------------- fused double-LIF -> binary spikes (f16) ----------------
__global__ void k_spikes(const float* __restrict__ x, const float* __restrict__ ssum,
                         const float* __restrict__ tsum, _Float16* __restrict__ m) {
  const int bn = blockIdx.x * blockDim.x + threadIdx.x;
  const int b  = bn >> 11;
  const float ts = tsum[bn];
  float vs = 0.f, vt = 0.f;
  #pragma unroll
  for (int t = 0; t < T_STEPS; ++t) {
    float xv = x[(size_t)t * BN + bn];
    float s1 = lif_step(vs, xv * ssum[t * B_DIM + b]);
    float s2 = lif_step(vt, xv * ts);
    m[(size_t)t * BN + bn] = (_Float16)(s1 * s2);
  }
}

// ---------------- W fp32 -> fp16 ----------------
__global__ void k_wconv(const float* __restrict__ W, _Float16* __restrict__ Wh) {
  const int i = (blockIdx.x * blockDim.x + threadIdx.x) * 4;
  #pragma unroll
  for (int j = 0; j < 4; ++j) Wh[i + j] = (_Float16)W[i + j];
}

// ---------------- TDM tile load: 128 rows x 32 halves, padded 80B LDS rows ----------------
// D# group0: count=1, lds_addr, global_addr(57b), type=2
// D# group1: data_size=2B, pad_enable, pad_interval=16 DWORDs, pad_amount=4 DWORDs,
//            tensor_dim0=2048, tensor_dim1=2048, tile 32x128, dim0_stride=2048
__device__ __forceinline__ void tdm_load_tile(const _Float16* gptr, unsigned lds_byte_off) {
  unsigned long long ga = (unsigned long long)(const void*)gptr;
  u32x4 g0 = { 1u,                                  // count = 1
               lds_byte_off,
               (unsigned)ga,
               (unsigned)(ga >> 32) | (2u << 30) }; // type = 2 (image)
  i32x8 g1 = { (int)((1u << 16) | (1u << 20) | (3u << 22) | (3u << 25)),
               (int)(2048u << 16),   // tensor_dim0[15:0] -> [63:48]
               (int)(2048u << 16),   // tensor_dim1[15:0] -> [95:80] ; dim0 hi = 0
               (int)(32u << 16),     // tile_dim0 = 32 -> [127:112] ; dim1 hi = 0
               128,                  // tile_dim1 = 128
               2048,                 // tensor_dim0_stride (low 32)
               0, 0 };
  i32x4 z4 = { 0, 0, 0, 0 };
  i32x8 z8 = { 0, 0, 0, 0, 0, 0, 0, 0 };
  __builtin_amdgcn_tensor_load_to_lds(g0, g1, z4, z4, z8, 0);
}

// ---------------- WMMA GEMM via TDM double buffering ----------------
// C[tb][mo] = sum_k m[tb][k] * Wh[mo][k]
// Block tile 128x128, 8 waves as 2x4, each wave 64x32 = 4x2 WMMA tiles.
// Dynamic LDS layout (bytes): [A0:0..10240) [B0:10240..20480) [A1:20480..30720) [B1:30720..40960)
__launch_bounds__(256)
__global__ void k_gemm(const _Float16* __restrict__ A, const _Float16* __restrict__ Wt,
                       float* __restrict__ C) {
  constexpr int K    = N_DIM;
  constexpr int LDA  = 40;              // 32 + 8 pad halves (80-byte rows, from TDM pad)
  constexpr int TILE = 128 * LDA;       // 5120 halves = 10240 bytes
  extern __shared__ _Float16 smem[];

  const int tid  = threadIdx.x;
  const int lane = tid & 31;
  const int wave = tid >> 5;

  const int brow = blockIdx.y * 128;    // tb base
  const int bcol = blockIdx.x * 128;    // mo base
  const int wrow = (wave >> 2) * 64;
  const int wcol = (wave & 3) * 32;

  // fragment gather indices per CDNA5 ISA VGPR layouts (wave32)
  const int mrow = lane & 15;
  const int kb_a = (lane < 16) ? 0 : 8;   // A: lanes0-15 K{0..7,16..23}, lanes16-31 K{8..15,24..31}
  const int ncol = lane & 15;
  const int kb_b = (lane < 16) ? 0 : 8;   // B: lanes0-15 K{0..7,16..23}, lanes16-31 K{8..15,24..31}

  v8f acc[4][2];
  #pragma unroll
  for (int i = 0; i < 4; ++i)
    #pragma unroll
    for (int j = 0; j < 2; ++j) { v8f z = {}; acc[i][j] = z; }

  // prologue: load buffer 0
  if (wave == 0) tdm_load_tile(A  + (size_t)brow * K, 0u);
  if (wave == 1) tdm_load_tile(Wt + (size_t)bcol * K, 10240u);

  constexpr int KSTEPS = K / 32;        // 64
  for (int kt = 0; kt < KSTEPS; ++kt) {
    const int cur = kt & 1;
    if (kt + 1 < KSTEPS) {
      const int nxt = cur ^ 1;
      const int k0n = (kt + 1) * 32;
      if (wave == 0) {
        tdm_load_tile(A + (size_t)brow * K + k0n, (unsigned)(nxt * 20480));
        __builtin_amdgcn_s_wait_tensorcnt(1);   // tile 'cur' resident (in-order per wave)
      }
      if (wave == 1) {
        tdm_load_tile(Wt + (size_t)bcol * K + k0n, (unsigned)(nxt * 20480 + 10240));
        __builtin_amdgcn_s_wait_tensorcnt(1);
      }
    } else {
      if (wave < 2) __builtin_amdgcn_s_wait_tensorcnt(0);
    }
    __syncthreads();                    // all waves see buffer 'cur'

    const _Float16* Asb = smem + cur * (2 * TILE);
    const _Float16* Bsb = Asb + TILE;

    v16h bf[2];
    #pragma unroll
    for (int j = 0; j < 2; ++j) {
      union { v16h h; v8h p[2]; } u;
      const _Float16* base = &Bsb[(wcol + j * 16 + ncol) * LDA + kb_b];
      u.p[0] = *(const v8h*)(base);
      u.p[1] = *(const v8h*)(base + 16);
      bf[j] = u.h;
    }
    #pragma unroll
    for (int i = 0; i < 4; ++i) {
      union { v16h h; v8h p[2]; } u;
      const _Float16* base = &Asb[(wrow + i * 16 + mrow) * LDA + kb_a];
      u.p[0] = *(const v8h*)(base);        // K kb..kb+7
      u.p[1] = *(const v8h*)(base + 16);   // K kb+16..kb+23
      v16h af = u.h;
      #pragma unroll
      for (int j = 0; j < 2; ++j) {
        acc[i][j] = __builtin_amdgcn_wmma_f32_16x16x32_f16(
            false, af, false, bf[j], (short)0, acc[i][j], false, false);
      }
    }
    __syncthreads();                    // buffer 'cur' free for next TDM overwrite
  }

  // C layout: VGPR v -> M = mbase + v, N = lane&15
  const int mbase = (lane < 16) ? 0 : 8;
  #pragma unroll
  for (int i = 0; i < 4; ++i)
    #pragma unroll
    for (int j = 0; j < 2; ++j) {
      const int c = bcol + wcol + j * 16 + ncol;
      #pragma unroll
      for (int v = 0; v < 8; ++v) {
        const int r = brow + wrow + i * 16 + mbase + v;
        C[(size_t)r * N_DIM + c] = acc[i][j][v];
      }
    }
}

// ---------------- BatchNorm stats ----------------
__global__ void k_bn_zero(float* p) { p[blockIdx.x * blockDim.x + threadIdx.x] = 0.f; }

__global__ void k_bn_partial(const float* __restrict__ g, float* __restrict__ bsum,
                             float* __restrict__ bsq) {
  const int c  = blockIdx.x * 256 + threadIdx.x;
  const int r0 = blockIdx.y * 128;
  float s = 0.f, s2 = 0.f;
  for (int r = r0; r < r0 + 128; ++r) {
    float v = g[(size_t)r * N_DIM + c];
    s += v; s2 += v * v;
  }
  atomicAdd(&bsum[c], s);
  atomicAdd(&bsq[c], s2);
}

__global__ void k_bn_final(const float* __restrict__ bsum, const float* __restrict__ bsq,
                           float* __restrict__ mean, float* __restrict__ rstd) {
  const int c = blockIdx.x * blockDim.x + threadIdx.x;
  float mu  = bsum[c] * (1.0f / TB);
  float var = bsq[c] * (1.0f / TB) - mu * mu;
  mean[c] = mu;
  rstd[c] = rsqrtf(var + 1e-5f);
}

// ---------------- final LIF + residual ----------------
__global__ void k_lif_final(const float* __restrict__ x, const float* __restrict__ g,
                            const float* __restrict__ mean, const float* __restrict__ rstd,
                            const float* __restrict__ gamma, const float* __restrict__ beta,
                            float* __restrict__ out) {
  const int bn = blockIdx.x * blockDim.x + threadIdx.x;
  const int b  = bn >> 11;
  const int n  = bn & (N_DIM - 1);
  const float mu = mean[n], rs = rstd[n], ga = gamma[n], be = beta[n];
  float v = 0.f;
  #pragma unroll
  for (int t = 0; t < T_STEPS; ++t) {
    float gv = g[(size_t)(t * B_DIM + b) * N_DIM + n];
    float nm = ga * (gv - mu) * rs + be;
    float s  = lif_step(v, nm);
    out[(size_t)t * BN + bn] = x[(size_t)t * BN + bn] + s;
  }
}

extern "C" void kernel_launch(void* const* d_in, const int* in_sizes, int n_in,
                              void* d_out, int out_size, void* d_ws, size_t ws_size,
                              hipStream_t stream) {
  (void)in_sizes; (void)n_in; (void)out_size; (void)ws_size;
  const float* x     = (const float*)d_in[0];
  const float* W     = (const float*)d_in[1];
  const float* gamma = (const float*)d_in[2];
  const float* beta  = (const float*)d_in[3];
  float* out = (float*)d_out;

  char* ws = (char*)d_ws;
  float*    ssum = (float*)   (ws + 0);           //   8 KB  [T*B]
  float*    tsum = (float*)   (ws + 8192);        // 512 KB  [B*N]
  _Float16* m    = (_Float16*)(ws + 532480);      //   8 MB  spikes f16 [TB][N]
  _Float16* Wh   = (_Float16*)(ws + 8921088);     //   8 MB  W f16
  float*    gem  = (float*)   (ws + 17309696);    //  16 MB  GEMM out [TB][N]
  float*    mean = (float*)   (ws + 34086912);
  float*    rstd = (float*)   (ws + 34095104);
  float*    bsum = (float*)   (ws + 34103296);
  float*    bsq  = (float*)   (ws + 34111488);

  k_spatial <<<TB,       256, 0, stream>>>(x, ssum);
  k_temporal<<<BN / 256, 256, 0, stream>>>(x, tsum);
  k_spikes  <<<BN / 256, 256, 0, stream>>>(x, ssum, tsum, m);
  k_wconv   <<<(N_DIM * N_DIM) / (256 * 4), 256, 0, stream>>>(W, Wh);

  dim3 gg(N_DIM / 128, TB / 128);
  k_gemm<<<gg, 256, 40960, stream>>>(m, Wh, gem);   // 40 KB dynamic LDS (2x double-buffered tiles)

  k_bn_zero<<<(2 * N_DIM) / 256, 256, 0, stream>>>(bsum);
  dim3 gp(N_DIM / 256, TB / 128);
  k_bn_partial<<<gp, 256, 0, stream>>>(gem, bsum, bsq);
  k_bn_final<<<N_DIM / 256, 256, 0, stream>>>(bsum, bsq, mean, rstd);

  k_lif_final<<<BN / 256, 256, 0, stream>>>(x, gem, mean, rstd, gamma, beta, out);
}
